// EditDecoderRNN_7181185319591
// MI455X (gfx1250) — matrix operand order repaired and checked by reading, not compiled, not run on hardware
//
#include <hip/hip_runtime.h>

typedef __attribute__((ext_vector_type(16))) _Float16 v16h;
typedef __attribute__((ext_vector_type(8)))  float    v8f;

#define DEV __device__ __forceinline__

// ---------------------------------------------------------------- WMMA core
DEV v8f wmma16(v16h a, v16h b, v8f c) {
  return __builtin_amdgcn_wmma_f32_16x16x32_f16(
      /*neg_a=*/false, a, /*neg_b=*/false, b,
      /*c_mod=*/(short)0, c, /*reuse_a=*/false, /*reuse_b=*/false);
}

// A fragment: 16x32 (MxK) f16, A row-major with leading dim lda.
// Contiguous-k layout used here: lanes 0-15 -> M=lane, K = k0+0..15;
//             lanes 16-31 -> M=lane-16, K = k0+16..31.
DEV v16h load_a_frag(const _Float16* A, int lda, int m0, int k0, int lane) {
  int row = m0 + (lane & 15);
  int kb  = k0 + ((lane >> 4) << 4);
  const _Float16* p = A + (size_t)row * lda + kb;
  v16h a;
#pragma unroll
  for (int i = 0; i < 16; ++i) { a[i] = p[i]; }
  return a;
}

// B fragment: 32x16 (KxN) where B[k][n] = W[n*ldw + k] (i.e. C = A @ W^T).
// lanes 0-15 -> N=lane, K=k0..k0+15; lanes 16-31 -> N=lane-16, K=k0+16..31.
DEV v16h load_b_wt(const _Float16* W, int ldw, int k0, int n0, int lane) {
  int col = n0 + (lane & 15);
  int kb  = k0 + ((lane >> 4) << 4);
  const _Float16* p = W + (size_t)col * ldw + kb;
  v16h b;
#pragma unroll
  for (int i = 0; i < 16; ++i) b[i] = p[i];
  return b;
}

// B fragment from a row-major KxN matrix: B[k][n] = P[k*ldb + n].
DEV v16h load_b_rm(const _Float16* P, int ldb, int k0, int n0, int lane) {
  int col = n0 + (lane & 15);
  int kb  = k0 + ((lane >> 4) << 4);
  v16h b;
#pragma unroll
  for (int i = 0; i < 16; ++i) b[i] = P[(size_t)(kb + i) * ldb + col];
  return b;
}

DEV float sigf(float x) { return 1.f / (1.f + expf(-x)); }

// ---------------------------------------------------------------- utilities
__global__ void k_f32_to_f16(const float* __restrict__ in, _Float16* __restrict__ out,
                             unsigned long long n) {
  unsigned long long i = (unsigned long long)blockIdx.x * blockDim.x + threadIdx.x;
  unsigned long long stride = (unsigned long long)gridDim.x * blockDim.x;
  for (; i < n; i += stride) out[i] = (_Float16)in[i];
}

__global__ void k_zero_f16(_Float16* __restrict__ p, unsigned long long n) {
  unsigned long long i = (unsigned long long)blockIdx.x * blockDim.x + threadIdx.x;
  unsigned long long stride = (unsigned long long)gridDim.x * blockDim.x;
  for (; i < n; i += stride) p[i] = (_Float16)0.f;
}

__global__ void k_add_bias(const float* __restrict__ a, const float* __restrict__ b,
                           float* __restrict__ o, int n) {
  int i = blockIdx.x * blockDim.x + threadIdx.x;
  if (i < n) o[i] = a[i] + b[i];
}

// x16[(t*16+b)*768 + k] = (f16) emb[ids[b*Tdim + t]*768 + k]
__global__ void k_gather_embed(const int* __restrict__ ids, const float* __restrict__ emb,
                               _Float16* __restrict__ x16, int Tdim, int nrows) {
  unsigned long long total = (unsigned long long)nrows * 768ull;
  unsigned long long i = (unsigned long long)blockIdx.x * blockDim.x + threadIdx.x;
  unsigned long long stride = (unsigned long long)gridDim.x * blockDim.x;
  for (; i < total; i += stride) {
    int m = (int)(i / 768), k = (int)(i % 768);
    int t = m >> 4, b = m & 15;
    int tok = ids[b * Tdim + t];
    x16[i] = (_Float16)emb[(size_t)tok * 768 + k];
  }
}

// cw16[(t*16+b)*768+k] = (f16) out_w[(ci[b,t]*16 + b)*768 + k]
__global__ void k_gather_cword(const int* __restrict__ ci, const float* __restrict__ out_w,
                               _Float16* __restrict__ cw16) {
  unsigned long long total = 2032ull * 768ull;
  unsigned long long i = (unsigned long long)blockIdx.x * blockDim.x + threadIdx.x;
  unsigned long long stride = (unsigned long long)gridDim.x * blockDim.x;
  for (; i < total; i += stride) {
    int m = (int)(i / 768), k = (int)(i % 768);
    int t = m >> 4, b = m & 15;
    int civ = ci[b * 127 + t];
    cw16[i] = (_Float16)out_w[((size_t)civ * 16 + b) * 768 + k];
  }
}

// ---------------------------------------------------------------- generic WMMA GEMM
DEV void gemm_epilogue(v8f acc, int m0, int col, int lane,
                       const float* __restrict__ bias, const float* __restrict__ madd,
                       float* __restrict__ out32, _Float16* __restrict__ out16,
                       int ldo, int rowmap, int nstore, int act) {
  float add = 0.f;
  if (col < nstore) {
    if (bias) add += bias[col];
    if (madd) add += madd[col];
  }
#pragma unroll
  for (int r = 0; r < 8; ++r) {
    int m = m0 + r + ((lane >> 4) << 3);
    float v = acc[r] + add;
    if (act == 1) v = tanhf(v);
    int orow;
    if (rowmap == 0) orow = m;
    else { int t = m >> 4, bb = m & 15; orow = (rowmap == 1) ? (bb * 128 + t) : (bb * 127 + t); }
    if (col < nstore) {
      if (out32) out32[(size_t)orow * ldo + col] = v;
      if (out16) out16[(size_t)orow * ldo + col] = (_Float16)v;
    }
  }
}

// D = act( A(M,K) @ W(N,K)^T + bias + madd ).  REQUIRES N % 64 == 0.
// Each wave owns FOUR 16x16 N-tiles sharing one A fragment; the k-loop is
// branch-free straight-line code (4 v_wmma per A load, accs pinned in regs).
// rowmap: 0 -> out row = m ; 1 -> (m=t*16+b) -> b*128+t ; 2 -> b*127+t
__global__ __launch_bounds__(256) void k_gemm_wt(
    const _Float16* __restrict__ A, int lda,
    const _Float16* __restrict__ W, int ldw,
    const float* __restrict__ bias, const float* __restrict__ madd,
    float* __restrict__ out32, _Float16* __restrict__ out16,
    int N, int K, int ldo, int rowmap, int nstore, int act) {
  int lane = threadIdx.x & 31, wave = threadIdx.x >> 5;
  int m0 = blockIdx.x * 16;
  int slot = blockIdx.y * 8 + wave;
  int n0 = slot * 64;                 // 4 tiles of 16 columns
  if (n0 >= N) return;                // wave-uniform

  v8f acc0 = {}, acc1 = {}, acc2 = {}, acc3 = {};
  const _Float16* aprefp = A + (size_t)(m0 + (lane & 15)) * lda;
  const _Float16* bprefp = W + (size_t)(n0 + (lane & 15)) * ldw;

  for (int k0 = 0; k0 < K; k0 += 32) {
    v16h a = load_a_frag(A, lda, m0, k0, lane);
    __builtin_prefetch((const void*)(aprefp + k0 + 512), 0, 1);   // global_prefetch_b8
    __builtin_prefetch((const void*)(bprefp + k0 + 512), 0, 1);
    acc0 = wmma16(a, load_b_wt(W, ldw, k0, n0,      lane), acc0);
    acc1 = wmma16(a, load_b_wt(W, ldw, k0, n0 + 16, lane), acc1);
    acc2 = wmma16(a, load_b_wt(W, ldw, k0, n0 + 32, lane), acc2);
    acc3 = wmma16(a, load_b_wt(W, ldw, k0, n0 + 48, lane), acc3);
  }

  int nc = n0 + (lane & 15);
  gemm_epilogue(acc0, m0, nc,      lane, bias, madd, out32, out16, ldo, rowmap, nstore, act);
  gemm_epilogue(acc1, m0, nc + 16, lane, bias, madd, out32, out16, ldo, rowmap, nstore, act);
  gemm_epilogue(acc2, m0, nc + 32, lane, bias, madd, out32, out16, ldo, rowmap, nstore, act);
  gemm_epilogue(acc3, m0, nc + 48, lane, bias, madd, out32, out16, ldo, rowmap, nstore, act);
}

// ---------------------------------------------------------------- LSTM recurrence
struct LstmArgs {
  const float*    xw[3];    // [steps][16][3072]  (x@Wih^T + bih + bhh)
  const _Float16* whh[3];   // [3072][768] f16 row-major
  float*          out[3];   // [steps][16][768]
  int             steps[3];
};

// One workgroup per LSTM. 768 threads = 24 waves; wave w owns hidden columns
// [32w, 32w+32): it computes the i/f/g/o gate tiles for exactly those columns
// (tiles tau = 48*g + 2*w + s), so the nonlinear update never crosses waves.
__global__ __launch_bounds__(768) void k_lstm(LstmArgs args) {
  int which = blockIdx.x;
  const float*    XW  = args.xw[which];
  const _Float16* Whh = args.whh[which];
  float*          out = args.out[which];
  int steps = args.steps[which];

  __shared__ _Float16 h16[16 * 768];
  int tid = threadIdx.x, lane = tid & 31, wave = tid >> 5;
  for (int i = tid; i < 16 * 768; i += 768) h16[i] = (_Float16)0.f;
  float cst[2][8];
#pragma unroll
  for (int s = 0; s < 2; ++s)
#pragma unroll
    for (int r = 0; r < 8; ++r) cst[s][r] = 0.f;
  __syncthreads();

  int mrow = (lane >> 4) << 3;
  int nidx = lane & 15;

  for (int t = 0; t < steps; ++t) {
    const float* XWt = XW + (size_t)t * 16 * 3072;
    v8f acc[4][2];
#pragma unroll
    for (int g = 0; g < 4; ++g)
#pragma unroll
      for (int s = 0; s < 2; ++s) {
        int n0 = 768 * g + 32 * wave + 16 * s;
        v8f c;
#pragma unroll
        for (int r = 0; r < 8; ++r)
          c[r] = XWt[(size_t)(r + mrow) * 3072 + n0 + nidx];
        acc[g][s] = c;
      }
    for (int k0 = 0; k0 < 768; k0 += 32) {
      v16h a = load_a_frag(h16, 768, 0, k0, lane);   // A from LDS
#pragma unroll
      for (int g = 0; g < 4; ++g)
#pragma unroll
        for (int s = 0; s < 2; ++s) {
          int n0 = 768 * g + 32 * wave + 16 * s;
          v16h b = load_b_wt(Whh, 768, k0, n0, lane);
          acc[g][s] = wmma16(a, b, acc[g][s]);
        }
    }
    float hval[2][8];
#pragma unroll
    for (int s = 0; s < 2; ++s)
#pragma unroll
      for (int r = 0; r < 8; ++r) {
        float iv = acc[0][s][r], fv = acc[1][s][r];
        float gv = acc[2][s][r], ov = acc[3][s][r];
        float c = sigf(fv) * cst[s][r] + sigf(iv) * tanhf(gv);
        float h = sigf(ov) * tanhf(c);
        cst[s][r] = c; hval[s][r] = h;
        int m = r + mrow;
        int col = 32 * wave + 16 * s + nidx;
        out[(size_t)t * 16 * 768 + (size_t)m * 768 + col] = h;
      }
    __syncthreads();   // all waves done reading h16 before we overwrite it
#pragma unroll
    for (int s = 0; s < 2; ++s)
#pragma unroll
      for (int r = 0; r < 8; ++r) {
        int m = r + mrow;
        int col = 32 * wave + 16 * s + nidx;
        h16[m * 768 + col] = (_Float16)hval[s][r];
      }
    __syncthreads();
  }
}

// ---------------------------------------------------------------- counters scan
__global__ void k_counters(const int* __restrict__ input_edits, const int* __restrict__ org_ids,
                           int* __restrict__ cd, int* __restrict__ ci, int* __restrict__ ca) {
  int b = threadIdx.x;
  if (b >= 16) return;
  int d = 0, i = 0, a = 0;
  for (int t = 0; t < 127; ++t) {
    cd[b * 127 + t] = d; ci[b * 127 + t] = i; ca[b * 127 + t] = a;
    int g = input_edits[b * 128 + t + 1];
    int d2 = d + (((g == 3) || (g == 4)) ? 1 : 0);
    bool alive = (g != 4) && (g != 2) && (g != 0);
    int i2 = i + (alive ? 1 : 0);
    int nxt = a + 1; if (nxt > 190) nxt = 190;
    int tok = org_ids[b * 191 + nxt];
    bool cond = alive && (g != 3) && ((a + 1) < 191) &&
                ((tok == 103) || (tok == 3) || (tok == 4));
    int mx = (d2 > a) ? d2 : a;
    int a2 = cond ? (a + 1) : mx;
    d = d2; i = i2; a = a2;
  }
}

// ---------------------------------------------------------------- attention
__global__ __launch_bounds__(256) void k_attn_scores(
    const _Float16* __restrict__ proj16, const _Float16* __restrict__ enc16,
    float* __restrict__ scores) {
  int lane = threadIdx.x & 31, wave = threadIdx.x >> 5;
  int b = blockIdx.x >> 3, mt = blockIdx.x & 7;
  int nt = blockIdx.y * 8 + wave;
  if (nt >= 12) return;
  int m0 = b * 128 + mt * 16, n0 = nt * 16;
  const _Float16* Wb = enc16 + (size_t)(b * 192 + 1) * 768;  // enc[:,1:] rows
  v8f acc = {};
  for (int k0 = 0; k0 < 768; k0 += 32) {
    v16h a = load_a_frag(proj16, 768, m0, k0, lane);
    v16h bb = load_b_wt(Wb, 768, k0, n0, lane);
    acc = wmma16(a, bb, acc);
  }
  int col = n0 + (lane & 15);
#pragma unroll
  for (int r = 0; r < 8; ++r) {
    int m = mt * 16 + r + ((lane >> 4) << 3);
    scores[((size_t)b * 128 + m) * 192 + col] = acc[r];
  }
}

__global__ __launch_bounds__(256) void k_attn_softmax(const float* __restrict__ scores,
                                                      _Float16* __restrict__ attn16) {
  __shared__ float red[256];
  int row = blockIdx.x, tid = threadIdx.x;
  const float* x = scores + (size_t)row * 192;
  float mx = -1e30f;
  for (int i = tid; i < 191; i += 256) mx = fmaxf(mx, x[i]);
  red[tid] = mx; __syncthreads();
  for (int s = 128; s > 0; s >>= 1) { if (tid < s) red[tid] = fmaxf(red[tid], red[tid + s]); __syncthreads(); }
  mx = red[0]; __syncthreads();
  float sm = 0.f;
  for (int i = tid; i < 191; i += 256) sm += expf(x[i] - mx);
  red[tid] = sm; __syncthreads();
  for (int s = 128; s > 0; s >>= 1) { if (tid < s) red[tid] += red[tid + s]; __syncthreads(); }
  float inv = 1.f / red[0];
  for (int i = tid; i < 192; i += 256) {
    float v = (i < 191) ? expf(x[i] - mx) * inv : 0.f;
    attn16[(size_t)row * 192 + i] = (_Float16)v;
  }
}

__global__ __launch_bounds__(256) void k_attn_apply(
    const _Float16* __restrict__ attn16, const _Float16* __restrict__ enc16,
    float* __restrict__ applied) {
  int lane = threadIdx.x & 31, wave = threadIdx.x >> 5;
  int b = blockIdx.x >> 3, mt = blockIdx.x & 7;
  int nt = blockIdx.y * 8 + wave;
  if (nt >= 48) return;
  int m0 = b * 128 + mt * 16, n0 = nt * 16;
  const _Float16* P = enc16 + (size_t)(b * 192 + 1) * 768;  // KxN row-major (l x h)
  v8f acc = {};
  for (int k0 = 0; k0 < 192; k0 += 32) {
    v16h a = load_a_frag(attn16, 192, m0, k0, lane);
    v16h bb = load_b_rm(P, 768, k0, n0, lane);
    acc = wmma16(a, bb, acc);
  }
  int col = n0 + (lane & 15);
#pragma unroll
  for (int r = 0; r < 8; ++r) {
    int m = m0 + r + ((lane >> 4) << 3);
    applied[(size_t)m * 768 + col] = acc[r];
  }
}

// ---------------------------------------------------------------- feature assembly
__global__ __launch_bounds__(256) void k_build_feats(
    const float* __restrict__ out_e, const float* __restrict__ out_a,
    const float* __restrict__ out_w, const float* __restrict__ applied,
    const float* __restrict__ enc32,
    const int* __restrict__ cd, const int* __restrict__ ci, const int* __restrict__ ca,
    const float* __restrict__ W_ref, const float* __restrict__ b_ref,
    _Float16* __restrict__ feat_e, _Float16* __restrict__ feat_a) {
  int lane = threadIdx.x & 31, wave = threadIdx.x >> 5;
  int wid = blockIdx.x * 8 + wave;
  if (wid >= 2032) return;
  int t = wid >> 4, b = wid & 15;
  const float* de  = out_e + (size_t)wid * 768;
  const float* da  = out_a + (size_t)wid * 768;
  int cdv = cd[b * 127 + t], civ = ci[b * 127 + t], cav = ca[b * 127 + t];
  const float* cin = enc32 + (size_t)(b * 192 + 1 + cdv) * 768;
  const float* can = enc32 + (size_t)(b * 192 + 1 + cav) * 768;
  const float* cw  = out_w + ((size_t)civ * 16 + b) * 768;
  const float* ap  = applied + ((size_t)b * 128 + t) * 768;

  float s0e = 0.f, s1e = 0.f, s0a = 0.f, s1a = 0.f;
  for (int h = lane; h < 768; h += 32) {
    float w0 = W_ref[h], w1 = W_ref[768 + h];
    float ve = de[h], va = da[h];
    s0e += ve * w0; s1e += ve * w1;
    s0a += va * w0; s1a += va * w1;
  }
  for (int off = 16; off > 0; off >>= 1) {
    s0e += __shfl_down(s0e, off, 32); s1e += __shfl_down(s1e, off, 32);
    s0a += __shfl_down(s0a, off, 32); s1a += __shfl_down(s1a, off, 32);
  }
  s0e = __shfl(s0e, 0, 32); s1e = __shfl(s1e, 0, 32);
  s0a = __shfl(s0a, 0, 32); s1a = __shfl(s1a, 0, 32);
  s0e += b_ref[0]; s1e += b_ref[1]; s0a += b_ref[0]; s1a += b_ref[1];

  float me = fmaxf(s0e, s1e), e0 = expf(s0e - me), e1 = expf(s1e - me);
  float we0 = e0 / (e0 + e1), we1 = e1 / (e0 + e1);
  float ma = fmaxf(s0a, s1a), a0 = expf(s0a - ma), a1 = expf(s1a - ma);
  float wa0 = a0 / (a0 + a1), wa1 = a1 / (a0 + a1);

  _Float16* fe = feat_e + (size_t)wid * 3840;
  _Float16* fa = feat_a + (size_t)wid * 3840;
  for (int h = lane; h < 768; h += 32) {
    float vde = de[h], vda = da[h], vap = ap[h];
    float vci = cin[h], vca = can[h], vcw = cw[h];
    fe[h]        = (_Float16)vde;
    fe[768 + h]  = (_Float16)vda;
    fe[1536 + h] = (_Float16)vap;
    fe[2304 + h] = (_Float16)(we0 * vci + we1 * vca);
    fe[3072 + h] = (_Float16)vcw;
    fa[h]        = (_Float16)vda;
    fa[768 + h]  = (_Float16)vde;
    fa[1536 + h] = (_Float16)vap;
    fa[2304 + h] = (_Float16)(wa0 * vci + wa1 * vca);
    fa[3072 + h] = (_Float16)vcw;
  }
}

// ---------------------------------------------------------------- log_softmax
// Row is staged once into LDS (30528 f32 = 122 KB << 320 KB WGP LDS), so HBM
// sees exactly one read + one write of the 248 MB logits instead of three reads.
__global__ __launch_bounds__(256) void k_logsoftmax_rows(float* __restrict__ x, int ld, int n) {
  __shared__ float buf[30528];
  __shared__ float red[256];
  float* row = x + (size_t)blockIdx.x * ld;
  int tid = threadIdx.x;
  for (int i = tid; i < n; i += 256) buf[i] = row[i];
  __syncthreads();
  float mx = -1e30f;
  for (int i = tid; i < n; i += 256) mx = fmaxf(mx, buf[i]);
  red[tid] = mx; __syncthreads();
  for (int s = 128; s > 0; s >>= 1) { if (tid < s) red[tid] = fmaxf(red[tid], red[tid + s]); __syncthreads(); }
  mx = red[0]; __syncthreads();
  float sm = 0.f;
  for (int i = tid; i < n; i += 256) sm += expf(buf[i] - mx);
  red[tid] = sm; __syncthreads();
  for (int s = 128; s > 0; s >>= 1) { if (tid < s) red[tid] += red[tid + s]; __syncthreads(); }
  float lg = logf(red[0]) + mx;
  for (int i = tid; i < n; i += 256) row[i] = buf[i] - lg;
}

// ---------------------------------------------------------------- host side
static void launch_gemm(hipStream_t s, const _Float16* A, int lda, const _Float16* W, int ldw,
                        const float* bias, const float* madd, float* o32, _Float16* o16,
                        int M, int N, int K, int ldo, int rowmap, int nstore, int act) {
  int slots = N / 64;                            // 4 N-tiles per wave; N % 64 == 0
  dim3 g(M / 16, (slots + 7) / 8);
  k_gemm_wt<<<g, 256, 0, s>>>(A, lda, W, ldw, bias, madd, o32, o16, N, K, ldo, rowmap, nstore, act);
}

extern "C" void kernel_launch(void* const* d_in, const int* in_sizes, int n_in,
                              void* d_out, int out_size, void* d_ws, size_t ws_size,
                              hipStream_t stream) {
  (void)in_sizes; (void)n_in; (void)out_size; (void)ws_size;

  const int*   input_edits   = (const int*)d_in[0];
  const int*   input_actions = (const int*)d_in[1];
  const int*   simp_sent     = (const int*)d_in[2];
  const int*   org_ids       = (const int*)d_in[3];
  const float* enc_in        = (const float*)d_in[4];
  const float* emb           = (const float*)d_in[5];
  const float* Wih_e = (const float*)d_in[6],  *Whh_e = (const float*)d_in[7];
  const float* bih_e = (const float*)d_in[8],  *bhh_e = (const float*)d_in[9];
  const float* Wih_a = (const float*)d_in[10], *Whh_a = (const float*)d_in[11];
  const float* bih_a = (const float*)d_in[12], *bhh_a = (const float*)d_in[13];
  const float* Wih_w = (const float*)d_in[14], *Whh_w = (const float*)d_in[15];
  const float* bih_w = (const float*)d_in[16], *bhh_w = (const float*)d_in[17];
  const float* W_align = (const float*)d_in[18];
  const float* W_proj  = (const float*)d_in[19];
  const float* W_ref   = (const float*)d_in[20];
  const float* b_ref   = (const float*)d_in[21];
  const float* W_mlp   = (const float*)d_in[22], *b_mlp = (const float*)d_in[23];
  const float* W_act   = (const float*)d_in[24], *b_act = (const float*)d_in[25];
  const float* W_out   = (const float*)d_in[26], *b_out = (const float*)d_in[27];
  const float* W_outact = (const float*)d_in[28], *b_outact = (const float*)d_in[29];
  const float* action_mask = (const float*)d_in[30];

  // ---- workspace bump allocator
  size_t off = 0;
  auto alloc = [&](size_t bytes) -> char* {
    off = (off + 255) & ~(size_t)255;
    char* p = (char*)d_ws + off;
    off += bytes;
    return p;
  };
  const size_t H2 = sizeof(_Float16);

  _Float16* x16_e   = (_Float16*)alloc(2048ull * 768 * H2);
  _Float16* x16_a   = (_Float16*)alloc(2048ull * 768 * H2);
  _Float16* x16_w   = (_Float16*)alloc(3072ull * 768 * H2);
  _Float16* Wih16_e = (_Float16*)alloc(3072ull * 768 * H2);
  _Float16* Wih16_a = (_Float16*)alloc(3072ull * 768 * H2);
  _Float16* Wih16_w = (_Float16*)alloc(3072ull * 768 * H2);
  _Float16* Whh16_e = (_Float16*)alloc(3072ull * 768 * H2);
  _Float16* Whh16_a = (_Float16*)alloc(3072ull * 768 * H2);
  _Float16* Whh16_w = (_Float16*)alloc(3072ull * 768 * H2);
  float*    bsum_e  = (float*)alloc(3072ull * 4);
  float*    bsum_a  = (float*)alloc(3072ull * 4);
  float*    bsum_w  = (float*)alloc(3072ull * 4);
  float*    XW_e    = (float*)alloc(2048ull * 3072 * 4);
  float*    XW_a    = (float*)alloc(2048ull * 3072 * 4);
  float*    XW_w    = (float*)alloc(3072ull * 3072 * 4);
  float*    out_e   = (float*)alloc(2048ull * 768 * 4);
  float*    out_a   = (float*)alloc(2048ull * 768 * 4);
  float*    out_w   = (float*)alloc(3072ull * 768 * 4);
  _Float16* encin16 = (_Float16*)alloc(3072ull * 768 * H2);
  _Float16* Walign16= (_Float16*)alloc(768ull * 768 * H2);
  float*    enc32   = (float*)alloc(3072ull * 768 * 4);
  _Float16* enc16   = (_Float16*)alloc(3088ull * 768 * H2);   // +16 zero pad rows
  int*      cd      = (int*)alloc(2032ull * 4);
  int*      ci      = (int*)alloc(2032ull * 4);
  int*      ca      = (int*)alloc(2032ull * 4);
  _Float16* cw16    = (_Float16*)alloc(2032ull * 768 * H2);
  _Float16* Wproj16 = (_Float16*)alloc(768ull * 768 * H2);
  _Float16* proj16  = (_Float16*)alloc(2048ull * 768 * H2);
  float*    scores  = (float*)alloc(2048ull * 192 * 4);
  _Float16* attn16  = (_Float16*)alloc(2048ull * 192 * H2);
  float*    applied = (float*)alloc(2048ull * 768 * 4);
  _Float16* feat_e  = (_Float16*)alloc(2032ull * 3840 * H2);
  _Float16* feat_a  = (_Float16*)alloc(2032ull * 3840 * H2);
  _Float16* Wmlp16  = (_Float16*)alloc(768ull * 3840 * H2);
  _Float16* Wact16  = (_Float16*)alloc(768ull * 3840 * H2);
  _Float16* hid_e   = (_Float16*)alloc(2032ull * 768 * H2);
  _Float16* hid_a   = (_Float16*)alloc(2032ull * 768 * H2);
  _Float16* Wout16  = (_Float16*)alloc(30528ull * 768 * H2);  // pad to mult of 64 cols
  _Float16* Woact16 = (_Float16*)alloc(256ull * 768 * H2);    // 200 -> 256 pad

  // ---- zero padded buffers
  k_zero_f16<<<1024, 256, 0, stream>>>(enc16,   3088ull * 768);
  k_zero_f16<<<1024, 256, 0, stream>>>(proj16,  2048ull * 768);
  k_zero_f16<<<4096, 256, 0, stream>>>(Wout16,  30528ull * 768);
  k_zero_f16<<<256,  256, 0, stream>>>(Woact16, 256ull * 768);

  // ---- f32 -> f16 conversions
  k_f32_to_f16<<<2048, 256, 0, stream>>>(Wih_e, Wih16_e, 3072ull * 768);
  k_f32_to_f16<<<2048, 256, 0, stream>>>(Whh_e, Whh16_e, 3072ull * 768);
  k_f32_to_f16<<<2048, 256, 0, stream>>>(Wih_a, Wih16_a, 3072ull * 768);
  k_f32_to_f16<<<2048, 256, 0, stream>>>(Whh_a, Whh16_a, 3072ull * 768);
  k_f32_to_f16<<<2048, 256, 0, stream>>>(Wih_w, Wih16_w, 3072ull * 768);
  k_f32_to_f16<<<2048, 256, 0, stream>>>(Whh_w, Whh16_w, 3072ull * 768);
  k_f32_to_f16<<<512,  256, 0, stream>>>(W_align, Walign16, 768ull * 768);
  k_f32_to_f16<<<512,  256, 0, stream>>>(W_proj,  Wproj16,  768ull * 768);
  k_f32_to_f16<<<2048, 256, 0, stream>>>(W_mlp, Wmlp16, 768ull * 3840);
  k_f32_to_f16<<<2048, 256, 0, stream>>>(W_act, Wact16, 768ull * 3840);
  k_f32_to_f16<<<4096, 256, 0, stream>>>(W_out, Wout16, 30522ull * 768);
  k_f32_to_f16<<<256,  256, 0, stream>>>(W_outact, Woact16, 200ull * 768);
  k_f32_to_f16<<<2048, 256, 0, stream>>>(enc_in, encin16, 3072ull * 768);
  k_add_bias<<<12, 256, 0, stream>>>(bih_e, bhh_e, bsum_e, 3072);
  k_add_bias<<<12, 256, 0, stream>>>(bih_a, bhh_a, bsum_a, 3072);
  k_add_bias<<<12, 256, 0, stream>>>(bih_w, bhh_w, bsum_w, 3072);

  // ---- embedding gathers, rows (t*16+b)
  k_gather_embed<<<1024, 256, 0, stream>>>(input_edits,   emb, x16_e, 128, 2048);
  k_gather_embed<<<1024, 256, 0, stream>>>(input_actions, emb, x16_a, 128, 2048);
  k_gather_embed<<<1024, 256, 0, stream>>>(simp_sent,     emb, x16_w, 192, 3072);

  // ---- batched input-gate GEMMs:  XW = x @ Wih^T + (bih + bhh)
  launch_gemm(stream, x16_e, 768, Wih16_e, 768, bsum_e, nullptr, XW_e, nullptr,
              2048, 3072, 768, 3072, 0, 3072, 0);
  launch_gemm(stream, x16_a, 768, Wih16_a, 768, bsum_a, nullptr, XW_a, nullptr,
              2048, 3072, 768, 3072, 0, 3072, 0);
  launch_gemm(stream, x16_w, 768, Wih16_w, 768, bsum_w, nullptr, XW_w, nullptr,
              3072, 3072, 768, 3072, 0, 3072, 0);

  // ---- three LSTM recurrences, one workgroup each, concurrent
  LstmArgs la;
  la.xw[0] = XW_e; la.whh[0] = Whh16_e; la.out[0] = out_e; la.steps[0] = 128;
  la.xw[1] = XW_a; la.whh[1] = Whh16_a; la.out[1] = out_a; la.steps[1] = 128;
  la.xw[2] = XW_w; la.whh[2] = Whh16_w; la.out[2] = out_w; la.steps[2] = 192;
  k_lstm<<<3, 768, 0, stream>>>(la);

  // ---- counters scan
  k_counters<<<1, 32, 0, stream>>>(input_edits, org_ids, cd, ci, ca);

  // ---- enc = tanh(encoder_outputs_org @ W_align^T)
  launch_gemm(stream, encin16, 768, Walign16, 768, nullptr, nullptr, enc32, enc16,
              3072, 768, 768, 768, 0, 768, 1);

  // ---- c_word gather + proj = c_word @ W_proj^T  (output in (b*128+t) rows)
  k_gather_cword<<<1024, 256, 0, stream>>>(ci, out_w, cw16);
  launch_gemm(stream, cw16, 768, Wproj16, 768, nullptr, nullptr, nullptr, proj16,
              2032, 768, 768, 768, 1, 768, 0);

  // ---- attention
  k_attn_scores<<<dim3(128, 2), 256, 0, stream>>>(proj16, enc16, scores);
  k_attn_softmax<<<2048, 256, 0, stream>>>(scores, attn16);
  k_attn_apply<<<dim3(128, 6), 256, 0, stream>>>(attn16, enc16, applied);

  // ---- feature assembly (wref softmax, counter gathers)
  k_build_feats<<<254, 256, 0, stream>>>(out_e, out_a, out_w, applied, enc32,
                                         cd, ci, ca, W_ref, b_ref, feat_e, feat_a);

  // ---- hidden MLPs: tanh(feat @ W^T + b)
  launch_gemm(stream, feat_e, 3840, Wmlp16, 3840, b_mlp, nullptr, nullptr, hid_e,
              2032, 768, 3840, 768, 0, 768, 1);
  launch_gemm(stream, feat_a, 3840, Wact16, 3840, b_act, nullptr, nullptr, hid_a,
              2032, 768, 3840, 768, 0, 768, 1);

  // ---- output logits straight into d_out (rows b*127+t)
  float* outp = (float*)d_out;
  const size_t EDIT = 2032ull * 30522ull;
  launch_gemm(stream, hid_e, 768, Wout16, 768, b_out, nullptr, outp, nullptr,
              2032, 30528, 768, 30522, 2, 30522, 0);
  launch_gemm(stream, hid_a, 768, Woact16, 768, b_outact, action_mask,
              outp + EDIT, nullptr, 2032, 256, 768, 200, 2, 200, 0);

  // ---- log_softmax in place
  k_logsoftmax_rows<<<2032, 256, 0, stream>>>(outp, 30522, 30522);
  k_logsoftmax_rows<<<2032, 256, 0, stream>>>(outp + EDIT, 200, 200);
}
